// InpaintGenerator_64922725646825
// MI455X (gfx1250) — matrix operand order repaired and checked by reading, not compiled, not run on hardware
//
#include <hip/hip_runtime.h>
#include <math.h>

#define BB   4
#define CC   256
#define HWp  16384
#define NH   8
#define HD   32
#define NR   256
#define REG  64
#define TOPK 4
#define KTOT 256            // TOPK * REG
#define SCALE_F 0.0625f     // 256^-0.5

#define PITCH32 40          // padded pitch for 32-K LDS tiles (80B rows)
#define PITCH256 264        // padded pitch for 256-K LDS tiles (528B rows)

typedef __attribute__((ext_vector_type(16))) __bf16 v16bf;
typedef __attribute__((ext_vector_type(8)))  float  v8f;

__device__ __forceinline__ v8f wmma_bf(const v16bf& a, const v16bf& b, const v8f& c) {
  return __builtin_amdgcn_wmma_f32_16x16x32_bf16(false, a, false, b, (short)0, c, false, false);
}

// fragment = two contiguous 8x bf16 runs -> two b128 loads
__device__ __forceinline__ v16bf load_frag2(const __bf16* p0, const __bf16* p1) {
  union { uint4 q[2]; v16bf v; } u;
  u.q[0] = *(const uint4*)p0;
  u.q[1] = *(const uint4*)p1;
  return u.v;
}

// async global->LDS copy, 16B per lane (CDNA5 GLOBAL_LOAD_ASYNC_TO_LDS_B128, ASYNCcnt)
__device__ __forceinline__ void async_copy_b128(__bf16* lds_dst, const __bf16* gsrc) {
  unsigned lds_addr = (unsigned)(size_t)lds_dst;            // low 32 bits = LDS offset
  unsigned long long ga = (unsigned long long)(size_t)gsrc;
  asm volatile("global_load_async_to_lds_b128 %0, %1, off"
               :: "v"(lds_addr), "v"(ga) : "memory");
}
__device__ __forceinline__ void wait_async0() {
  asm volatile("s_wait_asynccnt 0x0" ::: "memory");
}

// ---------------------------------------------------------------- K0: weights -> bf16 (vectorized)
__global__ void k_cvt_weights(const float* __restrict__ wq, const float* __restrict__ wo,
                              __bf16* __restrict__ wq_b, __bf16* __restrict__ wo_b) {
  int i = blockIdx.x * 256 + threadIdx.x;        // 4 elements per thread
  if (i < (768 * 256) / 4) {
    float4 v = ((const float4*)wq)[i];
    union { uint2 u; __bf16 h[4]; } pk;
    pk.h[0] = (__bf16)v.x; pk.h[1] = (__bf16)v.y;
    pk.h[2] = (__bf16)v.z; pk.h[3] = (__bf16)v.w;
    ((uint2*)wq_b)[i] = pk.u;
  }
  if (i < (256 * 256) / 4) {
    float4 v = ((const float4*)wo)[i];
    union { uint2 u; __bf16 h[4]; } pk;
    pk.h[0] = (__bf16)v.x; pk.h[1] = (__bf16)v.y;
    pk.h[2] = (__bf16)v.z; pk.h[3] = (__bf16)v.w;
    ((uint2*)wo_b)[i] = pk.u;
  }
}

// ---------------------------------------------------------------- K1: qkv = Wqkv * x  (per image)
__global__ __launch_bounds__(256) void k_qkv_gemm(
    const float* __restrict__ x, const __bf16* __restrict__ wq_b,
    const float* __restrict__ bias, float* __restrict__ qkv) {
  __shared__ __align__(16) __bf16 lds_a[64 * PITCH32];
  __shared__ __align__(16) __bf16 lds_bt[128 * PITCH32];
  const int tid = threadIdx.x;
  const int lane = tid & 31, wv = tid >> 5;
  const int wm = wv & 3, wn = wv >> 2;
  const int m0 = blockIdx.x * 64;
  const int b  = blockIdx.y >> 7;
  const int n0 = (blockIdx.y & 127) * 128;
  const float* xb = x + (size_t)b * CC * HWp;

  const int lm  = lane & 15;
  const int hi  = (lane & 16) ? 1 : 0;
  const int kb0 = hi * 8;    // A-fragment K half
  const int kbB = hi * 16;   // unused; B fragment also uses kb0

  const int arow = tid >> 2, acol = (tid & 3) * 8;  // A stage: 1 uint4 per thread
  const int kq = tid >> 5, nq = tid & 31;           // B stage: 4x4 transpose block

  v8f acc[4];
#pragma unroll
  for (int i = 0; i < 4; ++i) acc[i] = (v8f){0,0,0,0,0,0,0,0};

  for (int kk = 0; kk < CC; kk += 32) {
    // stage A (64x32 bf16): one b128 copy per thread
    *(uint4*)&lds_a[arow * PITCH32 + acol] =
        *(const uint4*)&wq_b[(size_t)(m0 + arow) * CC + kk + acol];
    // stage B (32x128 f32) -> bf16 transposed [n][k]
    float va[4][4];
#pragma unroll
    for (int j = 0; j < 4; ++j) {
      float4 t = *(const float4*)&xb[(size_t)(kk + kq * 4 + j) * HWp + n0 + nq * 4];
      va[j][0] = t.x; va[j][1] = t.y; va[j][2] = t.z; va[j][3] = t.w;
    }
#pragma unroll
    for (int i = 0; i < 4; ++i) {
      union { uint2 u; __bf16 h[4]; } pk;
#pragma unroll
      for (int j = 0; j < 4; ++j) pk.h[j] = (__bf16)va[j][i];
      *(uint2*)&lds_bt[(nq * 4 + i) * PITCH32 + kq * 4] = pk.u;
    }
    if (kk + 32 < CC)
      __builtin_prefetch(&xb[(size_t)(kk + 32 + kq * 4) * HWp + n0 + nq * 4], 0, 0);
    __syncthreads();

    const __bf16* ar = &lds_a[(wm * 16 + lm) * PITCH32];
    v16bf af = load_frag2(ar + kb0, ar + kb0 + 16);
#pragma unroll
    for (int nt = 0; nt < 4; ++nt) {
      const __bf16* br = &lds_bt[(wn * 64 + nt * 16 + lm) * PITCH32];
      v16bf bfr = load_frag2(br + kb0, br + kb0 + 16);
      acc[nt] = wmma_bf(af, bfr, acc[nt]);
    }
    __syncthreads();
  }
#pragma unroll
  for (int nt = 0; nt < 4; ++nt)
#pragma unroll
    for (int r = 0; r < 8; ++r) {
      int m = m0 + wm * 16 + r + hi * 8;
      int n = n0 + wn * 64 + nt * 16 + lm;
      qkv[((size_t)b * 768 + m) * HWp + n] = acc[nt][r] + bias[m];
    }
}

// ---------------------------------------------------------------- K2: grid->seq bf16
__global__ void k_seq_transform(const float* __restrict__ qkv,
                                __bf16* __restrict__ qs, __bf16* __restrict__ ks,
                                __bf16* __restrict__ vs) {
  int blk = blockIdx.x;            // b*8*256 + h*256 + r
  int r = blk & 255;
  int h = (blk >> 8) & 7;
  int b = blk >> 11;
  int ry = r >> 4, rx = r & 15;
  size_t dst = (size_t)blk * (REG * HD);
  for (int i = threadIdx.x; i < REG * HD; i += 256) {
    int t = i >> 5, d = i & 31;
    int y = ry * 8 + (t >> 3), xc = rx * 8 + (t & 7);
    size_t src = ((size_t)b * 768 + h * 32 + d) * HWp + (size_t)y * 128 + xc;
    qs[dst + i] = (__bf16)qkv[src];
    ks[dst + i] = (__bf16)qkv[src + (size_t)256 * HWp];
    vs[dst + i] = (__bf16)qkv[src + (size_t)512 * HWp];
  }
}

// ---------------------------------------------------------------- K3: region means
__global__ void k_region_mean(const float* __restrict__ qkv,
                              float* __restrict__ qr, float* __restrict__ kr) {
  int blk = blockIdx.x;            // b*256 + region
  int r = blk & 255, b = blk >> 8;
  int c = threadIdx.x;
  int ry = r >> 4, rx = r & 15;
  const float* qp = qkv + ((size_t)b * 768 + c) * HWp;
  const float* kp = qp + (size_t)256 * HWp;
  float sq = 0.f, sk = 0.f;
  for (int t = 0; t < 64; ++t) {
    size_t p = (size_t)(ry * 8 + (t >> 3)) * 128 + rx * 8 + (t & 7);
    sq += qp[p]; sk += kp[p];
  }
  qr[(size_t)blk * 256 + c] = sq * (1.0f / 64.0f);
  kr[(size_t)blk * 256 + c] = sk * (1.0f / 64.0f);
}

// ---------------------------------------------------------------- K4: routing + top-4
__global__ __launch_bounds__(256) void k_routing_topk(const float* __restrict__ qr,
                                                      const float* __restrict__ kr,
                                                      int* __restrict__ idx) {
  __shared__ float sv[256];
  __shared__ int   si[256];
  __shared__ int   winner;
  int blk = blockIdx.x;            // b*256 + query region
  int b = blk >> 8;
  int j = threadIdx.x;             // key region
  const float* qv = qr + (size_t)blk * 256;
  const float* kv = kr + ((size_t)b * 256 + j) * 256;
  float dot = 0.f;
  for (int c = 0; c < 256; ++c) dot += qv[c] * kv[c];
  float myv = dot;
  for (int kt = 0; kt < TOPK; ++kt) {
    sv[j] = myv; si[j] = j;
    __syncthreads();
    for (int s = 128; s > 0; s >>= 1) {
      if (j < s && sv[j + s] > sv[j]) { sv[j] = sv[j + s]; si[j] = si[j + s]; }
      __syncthreads();
    }
    if (j == 0) { winner = si[0]; idx[(size_t)blk * TOPK + kt] = si[0]; }
    __syncthreads();
    if (j == winner) myv = -3.0e38f;
    __syncthreads();
  }
}

// ---------------------------------------------------------------- K5: region attention (WMMA + async gather)
__global__ __launch_bounds__(128) void k_attention(
    const __bf16* __restrict__ qs, const __bf16* __restrict__ ks,
    const __bf16* __restrict__ vs, const int* __restrict__ idx,
    float* __restrict__ ao) {
  __shared__ __align__(16) __bf16 lds_kv[KTOT * PITCH32];   // phase1: K [key][32]p40 ; phase2: V^T [d][256]p264
  __shared__ __align__(16) __bf16 lds_p[REG * PITCH256];    // P [m][256]p264
  __shared__ int ridx[TOPK];

  const int tid = threadIdx.x;
  const int lane = tid & 31, wv = tid >> 5;
  const int blk = blockIdx.x;            // b*8*256 + h*256 + qr
  const int qr  = blk & 255;
  const int bh  = blk >> 8;              // b*8+h
  const int b   = bh >> 3;

  if (tid < TOPK) ridx[tid] = idx[((size_t)b * NR + qr) * TOPK + tid];
  __syncthreads();

  const __bf16* ksh   = ks + (size_t)bh * NR * REG * HD;
  const __bf16* vsh   = vs + (size_t)bh * NR * REG * HD;
  const __bf16* qtile = qs + (size_t)blk * (REG * HD);

  // stage gathered K rows [key][32] (pitch 40): async global->LDS b128 copies
  for (int i = tid; i < KTOT * 2; i += 128) {      // 512 x 16B transfers
    int key = i >> 2, part = i & 3;                // 4 uint4 per 32-dim row
    const __bf16* src = ksh + ((size_t)ridx[key >> 6] * REG + (key & 63)) * HD + part * 8;
    async_copy_b128(&lds_kv[key * PITCH32 + part * 8], src);
  }

  const int lm  = lane & 15;
  const int hi  = (lane & 16) ? 1 : 0;
  const int kb0 = hi * 8, kbB = hi * 16;

  // Q fragment straight from global (row-major [m][32]) — overlaps the async gather
  const __bf16* qrow = qtile + (wv * 16 + lm) * HD;
  v16bf af = load_frag2(qrow + kb0, qrow + kb0 + 16);

  wait_async0();
  __syncthreads();

  v8f acc[16];
#pragma unroll
  for (int nt = 0; nt < 16; ++nt) acc[nt] = (v8f){0,0,0,0,0,0,0,0};
#pragma unroll
  for (int nt = 0; nt < 16; ++nt) {
    const __bf16* kr = &lds_kv[(nt * 16 + lm) * PITCH32];
    v16bf bfr = load_frag2(kr + kb0, kr + kb0 + 16);
    acc[nt] = wmma_bf(af, bfr, acc[nt]);
  }

  // softmax: rows live in lane-halves; reduce 16 in-lane + xor over lanes 1,2,4,8
#pragma unroll
  for (int r = 0; r < 8; ++r) {
    float mx = -3.0e38f;
#pragma unroll
    for (int nt = 0; nt < 16; ++nt) mx = fmaxf(mx, acc[nt][r] * SCALE_F);
    mx = fmaxf(mx, __shfl_xor(mx, 1, 32));
    mx = fmaxf(mx, __shfl_xor(mx, 2, 32));
    mx = fmaxf(mx, __shfl_xor(mx, 4, 32));
    mx = fmaxf(mx, __shfl_xor(mx, 8, 32));
    float sum = 0.f;
#pragma unroll
    for (int nt = 0; nt < 16; ++nt) {
      float p = __expf(acc[nt][r] * SCALE_F - mx);
      acc[nt][r] = p;
      sum += p;
    }
    sum += __shfl_xor(sum, 1, 32);
    sum += __shfl_xor(sum, 2, 32);
    sum += __shfl_xor(sum, 4, 32);
    sum += __shfl_xor(sum, 8, 32);
    float inv = 1.0f / sum;
    int m = wv * 16 + r + hi * 8;
#pragma unroll
    for (int nt = 0; nt < 16; ++nt)
      lds_p[m * PITCH256 + nt * 16 + lm] = (__bf16)(acc[nt][r] * inv);
  }
  __syncthreads();

  // stage V transposed [d][key] (pitch 264) via register 4x16 transpose
  {
    const int kg = tid & 63, dg = tid >> 6;          // 4 keys x 16 dims per thread
    const __bf16* vreg = vsh + ((size_t)ridx[kg >> 4] * REG + ((kg * 4) & 63)) * HD + dg * 16;
    union { uint4 q[2]; __bf16 h[16]; } row[4];
#pragma unroll
    for (int j = 0; j < 4; ++j) {
      const uint4* s = (const uint4*)(vreg + j * HD);
      row[j].q[0] = s[0]; row[j].q[1] = s[1];
    }
#pragma unroll
    for (int i = 0; i < 16; ++i) {
      union { uint2 u; __bf16 h[4]; } pk;
#pragma unroll
      for (int j = 0; j < 4; ++j) pk.h[j] = row[j].h[i];
      *(uint2*)&lds_kv[(dg * 16 + i) * PITCH256 + kg * 4] = pk.u;
    }
  }
  __syncthreads();

  v8f oacc[2];
  oacc[0] = (v8f){0,0,0,0,0,0,0,0};
  oacc[1] = (v8f){0,0,0,0,0,0,0,0};
#pragma unroll
  for (int kk = 0; kk < KTOT; kk += 32) {
    const __bf16* pr = &lds_p[(wv * 16 + lm) * PITCH256 + kk];
    v16bf pf = load_frag2(pr + kb0, pr + kb0 + 16);
#pragma unroll
    for (int nt = 0; nt < 2; ++nt) {
      const __bf16* vr = &lds_kv[(nt * 16 + lm) * PITCH256 + kk];
      v16bf vf = load_frag2(vr + kb0, vr + kb0 + 16);
      oacc[nt] = wmma_bf(pf, vf, oacc[nt]);
    }
  }

  float* aop = ao + (size_t)blk * (REG * HD);
#pragma unroll
  for (int nt = 0; nt < 2; ++nt)
#pragma unroll
    for (int r = 0; r < 8; ++r) {
      int m = wv * 16 + r + hi * 8;
      aop[m * HD + nt * 16 + lm] = oacc[nt][r];
    }
}

// ---------------------------------------------------------------- K6: seq2grid + lepe + bias -> z(bf16)
__global__ void k_lepe_combine(const float* __restrict__ qkv, const float* __restrict__ ao,
                               const float* __restrict__ wl, const float* __restrict__ bl,
                               __bf16* __restrict__ z) {
  size_t gid = (size_t)blockIdx.x * 256 + threadIdx.x;    // over B*C*HW
  int p = (int)(gid & 16383);
  int c = (int)((gid >> 14) & 255);
  int b = (int)(gid >> 22);
  int y = p >> 7, xc = p & 127;
  int h = c >> 5, d = c & 31;
  int r = (y >> 3) * 16 + (xc >> 3);
  int t = (y & 7) * 8 + (xc & 7);
  float val = ao[(((size_t)(b * 8 + h) * NR + r) * REG + t) * HD + d];
  const float* vp = qkv + ((size_t)b * 768 + 512 + c) * HWp;
  const float* wc = wl + (size_t)c * 25;
  float acc = bl[c];
#pragma unroll
  for (int dy = -2; dy <= 2; ++dy) {
    int yy = y + dy;
    if (yy < 0 || yy >= 128) continue;
#pragma unroll
    for (int dx = -2; dx <= 2; ++dx) {
      int xx = xc + dx;
      if (xx < 0 || xx >= 128) continue;
      acc += wc[(dy + 2) * 5 + (dx + 2)] * vp[yy * 128 + xx];
    }
  }
  z[gid] = (__bf16)(val + acc);
}

// ---------------------------------------------------------------- K7: out = Wout * z + b_out
__global__ __launch_bounds__(256) void k_out_gemm(
    const __bf16* __restrict__ z, const __bf16* __restrict__ wo_b,
    const float* __restrict__ bias, float* __restrict__ out) {
  __shared__ __align__(16) __bf16 lds_a[64 * PITCH32];
  __shared__ __align__(16) __bf16 lds_bt[128 * PITCH32];
  const int tid = threadIdx.x;
  const int lane = tid & 31, wv = tid >> 5;
  const int wm = wv & 3, wn = wv >> 2;
  const int m0 = blockIdx.x * 64;
  const int b  = blockIdx.y >> 7;
  const int n0 = (blockIdx.y & 127) * 128;
  const __bf16* zb = z + (size_t)b * CC * HWp;

  const int lm  = lane & 15;
  const int hi  = (lane & 16) ? 1 : 0;
  const int kb0 = hi * 8, kbB = hi * 16;

  const int arow = tid >> 2, acol = (tid & 3) * 8;
  const int kq = tid >> 5, nq = tid & 31;

  v8f acc[4];
#pragma unroll
  for (int i = 0; i < 4; ++i) acc[i] = (v8f){0,0,0,0,0,0,0,0};

  for (int kk = 0; kk < CC; kk += 32) {
    *(uint4*)&lds_a[arow * PITCH32 + acol] =
        *(const uint4*)&wo_b[(size_t)(m0 + arow) * CC + kk + acol];
    // stage B (bf16 source) transposed [n][k]
    union { uint2 u; __bf16 h[4]; } tv[4];
#pragma unroll
    for (int j = 0; j < 4; ++j)
      tv[j].u = *(const uint2*)&zb[(size_t)(kk + kq * 4 + j) * HWp + n0 + nq * 4];
#pragma unroll
    for (int i = 0; i < 4; ++i) {
      union { uint2 u; __bf16 h[4]; } pk;
#pragma unroll
      for (int j = 0; j < 4; ++j) pk.h[j] = tv[j].h[i];
      *(uint2*)&lds_bt[(nq * 4 + i) * PITCH32 + kq * 4] = pk.u;
    }
    if (kk + 32 < CC)
      __builtin_prefetch(&zb[(size_t)(kk + 32 + kq * 4) * HWp + n0 + nq * 4], 0, 0);
    __syncthreads();

    const __bf16* ar = &lds_a[(wm * 16 + lm) * PITCH32];
    v16bf af = load_frag2(ar + kb0, ar + kb0 + 16);
#pragma unroll
    for (int nt = 0; nt < 4; ++nt) {
      const __bf16* br = &lds_bt[(wn * 64 + nt * 16 + lm) * PITCH32];
      v16bf bfr = load_frag2(br + kb0, br + kb0 + 16);
      acc[nt] = wmma_bf(af, bfr, acc[nt]);
    }
    __syncthreads();
  }
#pragma unroll
  for (int nt = 0; nt < 4; ++nt)
#pragma unroll
    for (int r = 0; r < 8; ++r) {
      int m = m0 + wm * 16 + r + hi * 8;
      int n = n0 + wn * 64 + nt * 16 + lm;
      out[((size_t)b * CC + m) * HWp + n] = acc[nt][r] + bias[m];
    }
}

// ---------------------------------------------------------------- launch
extern "C" void kernel_launch(void* const* d_in, const int* in_sizes, int n_in,
                              void* d_out, int out_size, void* d_ws, size_t ws_size,
                              hipStream_t stream) {
  (void)in_sizes; (void)n_in; (void)out_size; (void)ws_size;
  const float* x      = (const float*)d_in[0];
  const float* W_qkv  = (const float*)d_in[1];
  const float* b_qkv  = (const float*)d_in[2];
  const float* W_out  = (const float*)d_in[3];
  const float* b_out  = (const float*)d_in[4];
  const float* W_lepe = (const float*)d_in[5];
  const float* b_lepe = (const float*)d_in[6];
  float* out = (float*)d_out;

  char* ws = (char*)d_ws;
  size_t o = 0;
  float*  qkv  = (float*)(ws + o);  o += (size_t)BB * 768 * HWp * sizeof(float);
  __bf16* wq_b = (__bf16*)(ws + o); o += (size_t)768 * 256 * sizeof(__bf16);
  __bf16* wo_b = (__bf16*)(ws + o); o += (size_t)256 * 256 * sizeof(__bf16);
  __bf16* qs   = (__bf16*)(ws + o); o += (size_t)BB * NH * NR * REG * HD * sizeof(__bf16);
  __bf16* ks   = (__bf16*)(ws + o); o += (size_t)BB * NH * NR * REG * HD * sizeof(__bf16);
  __bf16* vs   = (__bf16*)(ws + o); o += (size_t)BB * NH * NR * REG * HD * sizeof(__bf16);
  float*  qrm  = (float*)(ws + o);  o += (size_t)BB * NR * CC * sizeof(float);
  float*  krm  = (float*)(ws + o);  o += (size_t)BB * NR * CC * sizeof(float);
  int*    idx  = (int*)(ws + o);    o += (size_t)BB * NR * TOPK * sizeof(int);
  o = (o + 255) & ~(size_t)255;
  float*  ao   = (float*)(ws + o);  o += (size_t)BB * NH * NR * REG * HD * sizeof(float);
  __bf16* zbuf = (__bf16*)(ws + o); o += (size_t)BB * CC * HWp * sizeof(__bf16);

  k_cvt_weights<<<192, 256, 0, stream>>>(W_qkv, W_out, wq_b, wo_b);
  k_qkv_gemm<<<dim3(12, 512), 256, 0, stream>>>(x, wq_b, b_qkv, qkv);
  k_seq_transform<<<BB * NH * NR, 256, 0, stream>>>(qkv, qs, ks, vs);
  k_region_mean<<<BB * NR, 256, 0, stream>>>(qkv, qrm, krm);
  k_routing_topk<<<BB * NR, 256, 0, stream>>>(qrm, krm, idx);
  k_attention<<<BB * NH * NR, 128, 0, stream>>>(qs, ks, vs, idx, ao);
  k_lepe_combine<<<(BB * CC * HWp) / 256, 256, 0, stream>>>(qkv, ao, W_lepe, b_lepe, zbuf);
  k_out_gemm<<<dim3(4, 512), 256, 0, stream>>>(zbuf, wo_b, b_out, out);
}
